// recurrent_noisy_42863773614477
// MI455X (gfx1250) — compile-verified, matrix operand order checked by reading, not run on hardware
//
#include <hip/hip_runtime.h>

// ---------------------------------------------------------------------------
// Problem dimensions (match the reference)
// ---------------------------------------------------------------------------
static constexpr int Bc = 128;
static constexpr int Tc = 512;
static constexpr int Ic = 512;
static constexpr int Hc = 1024;
static constexpr int Oc = 256;

typedef __attribute__((ext_vector_type(16))) __bf16 v16bf;
typedef __attribute__((ext_vector_type(8)))  float  v8f;
typedef __attribute__((ext_vector_type(4)))  float  v4f;

// ---------------------------------------------------------------------------
// Weight pack: fp32 row-major W[K][N] -> bf16, WMMA B-matrix (32x16 K-x-N tile)
// lane layout.  Packed layout is chosen so a wave loads one v16bf (32 bytes,
// two b128s) per B fragment:
//   tile (kt, nt), lane L, element j:
//     n  = nt*16 + (L & 15)
//     kb = kt*32 + ((L >> 4) * 8)
//     k  = (j < 8) ? kb + j : kb + 8 + j          //  {kb..kb+7} U {kb+16..kb+23}
//   stored at  P[((kt*NT16 + nt)*32 + L)*16 + j]  ==  P[idx] (identity)
// ---------------------------------------------------------------------------
__global__ void pack_b_kernel(const float* __restrict__ W, __bf16* __restrict__ P,
                              int K, int N) {
  int idx = blockIdx.x * blockDim.x + threadIdx.x;
  int total = K * N;
  if (idx >= total) return;
  int NT16 = N >> 4;
  int tIdx = idx >> 9;      // / 512 elements per 32x16 tile
  int r    = idx & 511;
  int lane = r >> 4;
  int j    = r & 15;
  int nt = tIdx % NT16;
  int kt = tIdx / NT16;
  int n  = nt * 16 + (lane & 15);
  int kb = kt * 32 + ((lane >> 4) << 3);
  int k  = kb + ((j < 8) ? j : (8 + j));
  P[idx] = (__bf16)W[(long)k * N + n];
}

// ---------------------------------------------------------------------------
// A fragment: fp32 row-major A (row stride lda), 16x32 tile at (m0, k0),
// converted to bf16 in registers using the ISA 16-bit A layout.
// ---------------------------------------------------------------------------
__device__ __forceinline__ v16bf load_a_frag_f32(const float* __restrict__ A,
                                                 long lda, int m0, int k0, int lane) {
  int row  = m0 + (lane & 15);
  int koff = k0 + ((lane >> 4) << 3);
  const float* p = A + (long)row * lda + koff;
  v4f a0 = *(const v4f*)(p + 0);
  v4f a1 = *(const v4f*)(p + 4);
  v4f a2 = *(const v4f*)(p + 16);
  v4f a3 = *(const v4f*)(p + 20);
  v16bf f;
  f[0]  = (__bf16)a0[0]; f[1]  = (__bf16)a0[1]; f[2]  = (__bf16)a0[2]; f[3]  = (__bf16)a0[3];
  f[4]  = (__bf16)a1[0]; f[5]  = (__bf16)a1[1]; f[6]  = (__bf16)a1[2]; f[7]  = (__bf16)a1[3];
  f[8]  = (__bf16)a2[0]; f[9]  = (__bf16)a2[1]; f[10] = (__bf16)a2[2]; f[11] = (__bf16)a2[3];
  f[12] = (__bf16)a3[0]; f[13] = (__bf16)a3[1]; f[14] = (__bf16)a3[2]; f[15] = (__bf16)a3[3];
  return f;
}

__device__ __forceinline__ v16bf load_b_frag(const __bf16* __restrict__ P,
                                             int NT16, int kt, int nt, int lane) {
  const __bf16* p = P + ((((long)kt * NT16 + nt) * 32 + lane) << 4);
  return *(const v16bf*)p;   // 32B contiguous per lane -> two global_load_b128
}

// ---------------------------------------------------------------------------
// Generic GEMM + bias:  C[M,N] = A[M,K](f32) * Bpack(bf16) + bias, f32 accum.
// One wave -> 16x64 output strip (A fragment reused across 4 WMMAs).
// blockDim.x == 256 (8 waves). M%16==0, N%64==0, K%32==0 required.
// ---------------------------------------------------------------------------
__global__ void __launch_bounds__(256)
gemm_bias_kernel(const float* __restrict__ A, long lda,
                 const __bf16* __restrict__ BP,
                 const float* __restrict__ bias,
                 float* __restrict__ C, long ldc,
                 int M, int N, int K) {
  int lane = threadIdx.x & 31;
  int wave = blockIdx.x * (blockDim.x >> 5) + (threadIdx.x >> 5);
  int ngroups = N >> 6;
  int mt = wave / ngroups;
  int ng = wave % ngroups;
  if (mt * 16 >= M) return;
  int m0 = mt * 16;
  int n0 = ng * 64;
  int NT16 = N >> 4;

  v8f zero = {0.f, 0.f, 0.f, 0.f, 0.f, 0.f, 0.f, 0.f};
  v8f acc0 = zero, acc1 = zero, acc2 = zero, acc3 = zero;

  int KT = K >> 5;
  for (int kt = 0; kt < KT; ++kt) {
    v16bf a  = load_a_frag_f32(A, lda, m0, kt << 5, lane);
    v16bf b0 = load_b_frag(BP, NT16, kt, (n0 >> 4) + 0, lane);
    v16bf b1 = load_b_frag(BP, NT16, kt, (n0 >> 4) + 1, lane);
    v16bf b2 = load_b_frag(BP, NT16, kt, (n0 >> 4) + 2, lane);
    v16bf b3 = load_b_frag(BP, NT16, kt, (n0 >> 4) + 3, lane);
    acc0 = __builtin_amdgcn_wmma_f32_16x16x32_bf16(false, a, false, b0, (short)0, acc0, false, false);
    acc1 = __builtin_amdgcn_wmma_f32_16x16x32_bf16(false, a, false, b1, (short)0, acc1, false, false);
    acc2 = __builtin_amdgcn_wmma_f32_16x16x32_bf16(false, a, false, b2, (short)0, acc2, false, false);
    acc3 = __builtin_amdgcn_wmma_f32_16x16x32_bf16(false, a, false, b3, (short)0, acc3, false, false);
  }

  // C/D layout: VGPR v -> row m0 + v + (lane>=16 ? 8 : 0); col n0 + 16*i + (lane&15)
  int ncol = lane & 15;
  int mrow = m0 + ((lane >> 4) << 3);
  v8f accs[4] = {acc0, acc1, acc2, acc3};
#pragma unroll
  for (int i = 0; i < 4; ++i) {
    int n = n0 + i * 16 + ncol;
    float bv = bias[n];
#pragma unroll
    for (int v = 0; v < 8; ++v) {
      C[(long)(mrow + v) * ldc + n] = accs[i][v] + bv;
    }
  }
}

// ---------------------------------------------------------------------------
// One recurrent step:  h_t = relu(xp_t + h_{t-1} @ W_hh + b_hh + sigma * n_t)
// xp_t is read from net_units[:,t,:] (written there by GEMM-1) and
// overwritten in place with h_t.  M=128, N=K=1024 fixed.
// ---------------------------------------------------------------------------
__global__ void __launch_bounds__(256)
rnn_step_kernel(const float* __restrict__ Hprev, long ldh,
                const __bf16* __restrict__ WhhP,
                const float* __restrict__ bias,
                const float* __restrict__ noise_t, long ldnz,
                const float* __restrict__ sigma_p,
                float* __restrict__ net_t, long ldn) {
  constexpr int N = Hc, K = Hc;
  int lane = threadIdx.x & 31;
  int wave = blockIdx.x * (blockDim.x >> 5) + (threadIdx.x >> 5);
  constexpr int ngroups = N >> 6;     // 16
  int mt = wave / ngroups;
  int ng = wave % ngroups;
  int m0 = mt * 16;
  int n0 = ng * 64;
  constexpr int NT16 = N >> 4;
  float sigma = *sigma_p;

  v8f zero = {0.f, 0.f, 0.f, 0.f, 0.f, 0.f, 0.f, 0.f};
  v8f acc0 = zero, acc1 = zero, acc2 = zero, acc3 = zero;

  for (int kt = 0; kt < (K >> 5); ++kt) {
    v16bf a  = load_a_frag_f32(Hprev, ldh, m0, kt << 5, lane);
    v16bf b0 = load_b_frag(WhhP, NT16, kt, (n0 >> 4) + 0, lane);
    v16bf b1 = load_b_frag(WhhP, NT16, kt, (n0 >> 4) + 1, lane);
    v16bf b2 = load_b_frag(WhhP, NT16, kt, (n0 >> 4) + 2, lane);
    v16bf b3 = load_b_frag(WhhP, NT16, kt, (n0 >> 4) + 3, lane);
    acc0 = __builtin_amdgcn_wmma_f32_16x16x32_bf16(false, a, false, b0, (short)0, acc0, false, false);
    acc1 = __builtin_amdgcn_wmma_f32_16x16x32_bf16(false, a, false, b1, (short)0, acc1, false, false);
    acc2 = __builtin_amdgcn_wmma_f32_16x16x32_bf16(false, a, false, b2, (short)0, acc2, false, false);
    acc3 = __builtin_amdgcn_wmma_f32_16x16x32_bf16(false, a, false, b3, (short)0, acc3, false, false);
  }

  int ncol = lane & 15;
  int mrow = m0 + ((lane >> 4) << 3);
  v8f accs[4] = {acc0, acc1, acc2, acc3};
#pragma unroll
  for (int i = 0; i < 4; ++i) {
    int n = n0 + i * 16 + ncol;
    float bv = bias[n];
#pragma unroll
    for (int v = 0; v < 8; ++v) {
      int m = mrow + v;                              // batch index
      long idx = (long)m * ldn + n;
      float xp = net_t[idx];                         // pre-projected input
      float nz = noise_t[(long)m * ldnz + n];
      float h  = accs[i][v] + xp + bv + sigma * nz;
      net_t[idx] = h > 0.f ? h : 0.f;                // relu, overwrite in place
    }
  }
}

// ---------------------------------------------------------------------------
// Strided slice copy: dst[r*cols + c] = src[r*srcRowStride + c]
// ---------------------------------------------------------------------------
__global__ void copy_last_kernel(const float* __restrict__ src, float* __restrict__ dst,
                                 int rows, int cols, long srcRowStride) {
  int i = blockIdx.x * blockDim.x + threadIdx.x;
  if (i >= rows * cols) return;
  int r = i / cols;
  int c = i - r * cols;
  dst[i] = src[(long)r * srcRowStride + c];
}

// ---------------------------------------------------------------------------
// Host-side orchestration (all launches on `stream`, graph-capture safe)
// ---------------------------------------------------------------------------
extern "C" void kernel_launch(void* const* d_in, const int* in_sizes, int n_in,
                              void* d_out, int out_size, void* d_ws, size_t ws_size,
                              hipStream_t stream) {
  const float* input  = (const float*)d_in[0];   // [B,T,I]
  const float* noise  = (const float*)d_in[1];   // [B,T,H]
  const float* hidden = (const float*)d_in[2];   // [B,H]
  const float* sigma  = (const float*)d_in[3];   // scalar (device)
  const float* W_in   = (const float*)d_in[4];   // [I,H]
  const float* b_in   = (const float*)d_in[5];   // [H]
  const float* W_hh   = (const float*)d_in[6];   // [H,H]
  const float* b_hh   = (const float*)d_in[7];   // [H]
  const float* W_fc   = (const float*)d_in[8];   // [H,O]
  const float* b_fc   = (const float*)d_in[9];   // [O]

  // d_out = out[B,O] | h_last[B,H] | net_units[B,T,H] | read_out_units[B,T,O]
  float* out_p   = (float*)d_out;
  float* hlast_p = out_p + (long)Bc * Oc;
  float* net_p   = hlast_p + (long)Bc * Hc;
  float* rou_p   = net_p + (long)Bc * Tc * Hc;

  // workspace: bf16-packed weights (~3.5 MB)
  __bf16* WinP = (__bf16*)d_ws;
  __bf16* WhhP = WinP + (long)Ic * Hc;
  __bf16* WfcP = WhhP + (long)Hc * Hc;

  // 1) pack weights to WMMA-friendly bf16 layout
  {
    int n1 = Ic * Hc;
    pack_b_kernel<<<(n1 + 255) / 256, 256, 0, stream>>>(W_in, WinP, Ic, Hc);
    int n2 = Hc * Hc;
    pack_b_kernel<<<(n2 + 255) / 256, 256, 0, stream>>>(W_hh, WhhP, Hc, Hc);
    int n3 = Hc * Oc;
    pack_b_kernel<<<(n3 + 255) / 256, 256, 0, stream>>>(W_fc, WfcP, Hc, Oc);
  }

  // 2) xp = input @ W_in + b_in  -> written directly into net_units slot
  {
    int M = Bc * Tc, N = Hc, K = Ic;
    int waves = (M / 16) * (N / 64);          // 65536
    gemm_bias_kernel<<<waves / 8, 256, 0, stream>>>(
        input, (long)Ic, WinP, b_in, net_p, (long)Hc, M, N, K);
  }

  // 3) recurrence: 512 stream-ordered steps (stream order = global sync)
  for (int t = 0; t < Tc; ++t) {
    const float* Hprev = (t == 0) ? hidden : (net_p + (long)(t - 1) * Hc);
    long ldh = (t == 0) ? (long)Hc : (long)Tc * Hc;
    int waves = (Bc / 16) * (Hc / 64);        // 128 waves -> 16 blocks
    rnn_step_kernel<<<waves / 8, 256, 0, stream>>>(
        Hprev, ldh, WhhP, b_hh,
        noise + (long)t * Hc, (long)Tc * Hc,
        sigma,
        net_p + (long)t * Hc, (long)Tc * Hc);
  }

  // 4) read_out_units = net_units @ W_fc + b_fc
  {
    int M = Bc * Tc, N = Oc, K = Hc;
    int waves = (M / 16) * (N / 64);          // 16384
    gemm_bias_kernel<<<waves / 8, 256, 0, stream>>>(
        net_p, (long)Hc, WfcP, b_fc, rou_p, (long)Oc, M, N, K);
  }

  // 5) h_last = net_units[:, T-1, :];  out = read_out_units[:, T-1, :]
  copy_last_kernel<<<((Bc * Hc) + 255) / 256, 256, 0, stream>>>(
      net_p + (long)(Tc - 1) * Hc, hlast_p, Bc, Hc, (long)Tc * Hc);
  copy_last_kernel<<<((Bc * Oc) + 255) / 256, 256, 0, stream>>>(
      rou_p + (long)(Tc - 1) * Oc, out_p, Bc, Oc, (long)Tc * Oc);
}
